// TSAGEConv_1855425871960
// MI455X (gfx1250) — matrix operand order repaired and checked by reading, not compiled, not run on hardware
//
#include <hip/hip_runtime.h>

// ---- problem constants (match reference) ----
constexpr int E_EDGES = 262144;
constexpr int BK      = 8192;
constexpr int DDEG    = 32;
constexpr int DIM     = 128;
constexpr int OUTF    = 128;
constexpr int K1      = 256;   // encode GEMM K (concat [u | time])
constexpr int K2      = 256;   // conv GEMM K (concat [self | mean-neigh])

typedef float  v8f  __attribute__((ext_vector_type(8)));
typedef __bf16 v16bf __attribute__((ext_vector_type(16)));
typedef __bf16 v8bf  __attribute__((ext_vector_type(8)));

// ---- fp32 <-> bf16 split helpers (RNE) ----
__device__ __forceinline__ unsigned short f2bf(float f) {
    unsigned u = __float_as_uint(f);
    u += 0x7FFFu + ((u >> 16) & 1u);
    return (unsigned short)(u >> 16);
}
__device__ __forceinline__ float bf2f(unsigned short h) {
    return __uint_as_float((unsigned)h << 16);
}

// A-fragment (16-bit A 16x32): lane l holds M=l&15; r=l>>4 selects
// K run [k0+16r .. +15] (contiguous, matching the B-fragment).
__device__ __forceinline__ v16bf ld_afrag(const unsigned short* row, int k0, int r) {
    const v8bf a0 = *reinterpret_cast<const v8bf*>(row + k0 + 16 * r);
    const v8bf a1 = *reinterpret_cast<const v8bf*>(row + k0 + 16 * r + 8);
    return __builtin_shufflevector(a0, a1, 0,1,2,3,4,5,6,7,8,9,10,11,12,13,14,15);
}
// B-fragment (16-bit B 32x16): lane l holds col n=l&15, K = k0+16*(l>>4) .. +15,
// i.e. 16 consecutive bf16 of W[n][k] -> one 32-byte contiguous load.
__device__ __forceinline__ v16bf ld_bfrag(const unsigned short* W, int ldw, int col, int kb) {
    return *reinterpret_cast<const v16bf*>(W + (size_t)col * ldw + kb);
}

// bf16x3 split-precision MAC: A*B ~= Ah*Bh + Ah*Bl + Al*Bh (fp32 accumulate)
__device__ __forceinline__ v8f wmma_bf16x3(v16bf ah, v16bf al, v16bf bh, v16bf bl, v8f c) {
    c = __builtin_amdgcn_wmma_f32_16x16x32_bf16(false, ah, false, bh, (short)0, c, false, false);
    c = __builtin_amdgcn_wmma_f32_16x16x32_bf16(false, ah, false, bl, (short)0, c, false, false);
    c = __builtin_amdgcn_wmma_f32_16x16x32_bf16(false, al, false, bh, (short)0, c, false, false);
    return c;
}

// -------------------------------------------------------------------------
// Kernel 0: split fp32 weights into hi/lo bf16 arrays (run once, tiny).
// -------------------------------------------------------------------------
__global__ __launch_bounds__(256) void cvt_hilo(
    const float* __restrict__ in, unsigned short* __restrict__ hi,
    unsigned short* __restrict__ lo, int n)
{
    const int i = blockIdx.x * 256 + threadIdx.x;
    if (i < n) {
        const float x = in[i];
        const unsigned short h = f2bf(x);
        hi[i] = h;
        lo[i] = f2bf(x - bf2f(h));
    }
}

// -------------------------------------------------------------------------
// Kernel 1: time-encode.  enc = relu([u | cos(t*freq+phase)] @ W_time^T + b)
// block = 128 threads (4 waves), 16 edges x 128 out cols; grid.y: src/dst.
// -------------------------------------------------------------------------
__global__ __launch_bounds__(128) void tsage_encode(
    const float* __restrict__ src_feat, const float* __restrict__ dst_feat,
    const float* __restrict__ timestamp,
    const float* __restrict__ basis_freq, const float* __restrict__ phase,
    const unsigned short* __restrict__ Wt_hi, const unsigned short* __restrict__ Wt_lo,
    const float* __restrict__ b_time,
    float* __restrict__ enc_src, float* __restrict__ enc_dst)
{
    constexpr int TS   = 16;
    constexpr int LDXB = K1 + 8;                         // 264 bf16: pad rows
    __shared__ __align__(16) unsigned short XhiS[TS * LDXB];   // 8.4 KB
    __shared__ __align__(16) unsigned short XloS[TS * LDXB];   // 8.4 KB

    const int tid = threadIdx.x;
    const int m0  = blockIdx.x * TS;
    const int grp = blockIdx.y;
    const float* u    = grp ? dst_feat : src_feat;
    float*       encO = grp ? enc_dst  : enc_src;

    // build X = [u | cos(t*freq + phase)], split hi/lo at write time
    for (int idx = tid; idx < TS * K1; idx += 128) {
        const int m = idx >> 8;
        const int k = idx & 255;
        float v;
        if (k < DIM) {
            v = u[(size_t)(m0 + m) * DIM + k];
        } else {
            const int j = k - DIM;
            v = cosf(timestamp[m0 + m] * basis_freq[j] + phase[j]);
        }
        const unsigned short h = f2bf(v);
        XhiS[m * LDXB + k] = h;
        XloS[m * LDXB + k] = f2bf(v - bf2f(h));
    }
    __syncthreads();

    const int lane = tid & 31;
    const int wave = tid >> 5;
    const int r    = lane >> 4;
    const int mloc = lane & 15;
    const int col0 = wave * 32 + mloc;       // wave owns cols [wave*32, wave*32+31]

    v8f acc0 = {}; v8f acc1 = {};
    const unsigned short* rowHi = &XhiS[mloc * LDXB];
    const unsigned short* rowLo = &XloS[mloc * LDXB];

    __builtin_prefetch(Wt_hi + (size_t)col0 * K1, 0, 1);        // L2-resident weights
    __builtin_prefetch(Wt_hi + (size_t)(col0 + 16) * K1, 0, 1);

    for (int k0 = 0; k0 < K1; k0 += 32) {
        const int kb = k0 + 16 * r;
        const v16bf ah  = ld_afrag(rowHi, k0, r);
        const v16bf al  = ld_afrag(rowLo, k0, r);
        const v16bf bh0 = ld_bfrag(Wt_hi, K1, col0, kb);
        const v16bf bl0 = ld_bfrag(Wt_lo, K1, col0, kb);
        const v16bf bh1 = ld_bfrag(Wt_hi, K1, col0 + 16, kb);
        const v16bf bl1 = ld_bfrag(Wt_lo, K1, col0 + 16, kb);
        acc0 = wmma_bf16x3(ah, al, bh0, bl0, acc0);
        acc1 = wmma_bf16x3(ah, al, bh1, bl1, acc1);
    }

    const float bias0 = b_time[col0];
    const float bias1 = b_time[col0 + 16];
    #pragma unroll
    for (int i = 0; i < 8; ++i) {
        const int m = i + 8 * r;             // C/D layout: VGPR i -> rows i / i+8
        encO[(size_t)(m0 + m) * DIM + col0]      = fmaxf(acc0[i] + bias0, 0.0f);
        encO[(size_t)(m0 + m) * DIM + col0 + 16] = fmaxf(acc1[i] + bias1, 0.0f);
    }
}

// -------------------------------------------------------------------------
// Kernel 2: per-bucket causal mean conv + fused dual GEMM + scatter.
// block = 128 threads (4 waves), one bucket of D=32 edges; grid.y: grouping.
// -------------------------------------------------------------------------
__global__ __launch_bounds__(128) void tsage_meanconv(
    const float* __restrict__ enc_src, const float* __restrict__ enc_dst,
    const int* __restrict__ src_perm, const int* __restrict__ dst_perm,
    const int* __restrict__ src_deg, const int* __restrict__ dst_deg,
    const unsigned short* __restrict__ Ws_hi, const unsigned short* __restrict__ Ws_lo,
    const unsigned short* __restrict__ Wn_hi, const unsigned short* __restrict__ Wn_lo,
    const float* __restrict__ b_self, const float* __restrict__ b_neigh,
    float* __restrict__ out)                  // [2][E][OUT] flat
{
    constexpr int LDXB = K2 + 8;              // 264 bf16
    constexpr int LDH  = DIM + 4;             // 132 f32
    __shared__ __align__(16) unsigned short XhiS[DDEG * LDXB];  // 16.9 KB
    __shared__ __align__(16) unsigned short XloS[DDEG * LDXB];  // 16.9 KB
    __shared__ float Hn[DDEG * LDH];                            // 16.9 KB
    __shared__ int   permS[DDEG];
    __shared__ int   idxS[DDEG];

    const int tid = threadIdx.x;
    const int bkt = blockIdx.x;
    const int grp = blockIdx.y;
    const float* encSelf = grp ? enc_dst : enc_src;
    const float* encNei  = grp ? enc_src : enc_dst;
    const int*   perm    = (grp ? dst_perm : src_perm) + bkt * DDEG;
    const int*   degi    = (grp ? dst_deg  : src_deg ) + bkt * DDEG;
    float*       outG    = out + (size_t)grp * E_EDGES * OUTF;

    if (tid < DDEG) { permS[tid] = perm[tid]; idxS[tid] = degi[tid]; }
    __syncthreads();

    // gather: self -> X[:,0:128] (split hi/lo), neigh -> Hn (fp32 for cumsum)
    for (int idx = tid; idx < DDEG * DIM; idx += 128) {
        const int m = idx >> 7;
        const int k = idx & 127;
        const size_t e = (size_t)permS[m];
        const float s = encSelf[e * DIM + k];
        const unsigned short h = f2bf(s);
        XhiS[m * LDXB + k] = h;
        XloS[m * LDXB + k] = f2bf(s - bf2f(h));
        Hn[m * LDH + k] = encNei[e * DIM + k];
    }
    __syncthreads();

    // column-parallel causal cumsum (128 threads = 128 dims, fp32 exact order)
    {
        const int k = tid;
        float run = 0.0f;
        #pragma unroll
        for (int m = 0; m < DDEG; ++m) {
            run += Hn[m * LDH + k];
            Hn[m * LDH + k] = run;
        }
    }
    __syncthreads();

    // causal mean gather -> X[:,128:256], split hi/lo
    {
        const int k = tid;
        for (int m = 0; m < DDEG; ++m) {
            const int id = idxS[m];
            const float v = Hn[id * LDH + k] / (float)(id + 1);
            const unsigned short h = f2bf(v);
            XhiS[m * LDXB + DIM + k] = h;
            XloS[m * LDXB + DIM + k] = f2bf(v - bf2f(h));
        }
    }
    __syncthreads();

    // GEMM: [32 x 256] @ [256 x 128]; K<128 -> W_self, K>=128 -> W_neigh.
    const int lane = tid & 31;
    const int wave = tid >> 5;
    const int r    = lane >> 4;
    const int mloc = lane & 15;
    const int mb   = (wave & 1) * 16;         // row tile
    const int nb   = (wave >> 1) * 64;        // 4 col tiles per wave

    v8f acc[4] = {};
    const unsigned short* rowHi = &XhiS[(mb + mloc) * LDXB];
    const unsigned short* rowLo = &XloS[(mb + mloc) * LDXB];

    __builtin_prefetch(Ws_hi + (size_t)(nb + mloc) * DIM, 0, 1);
    __builtin_prefetch(Wn_hi + (size_t)(nb + mloc) * DIM, 0, 1);

    for (int k0 = 0; k0 < K2; k0 += 32) {     // k-range uniform per half (128%32==0)
        const unsigned short* Whi = (k0 < DIM) ? Ws_hi : Wn_hi;
        const unsigned short* Wlo = (k0 < DIM) ? Ws_lo : Wn_lo;
        const int kw = k0 & (DIM - 1);
        const int kb = kw + 16 * r;
        const v16bf ah = ld_afrag(rowHi, k0, r);
        const v16bf al = ld_afrag(rowLo, k0, r);
        #pragma unroll
        for (int j = 0; j < 4; ++j) {
            const int col = nb + j * 16 + mloc;
            const v16bf bh = ld_bfrag(Whi, DIM, col, kb);
            const v16bf bl = ld_bfrag(Wlo, DIM, col, kb);
            acc[j] = wmma_bf16x3(ah, al, bh, bl, acc[j]);
        }
    }

    // bias + scatter back to flat edge order (perm is a permutation: full cover)
    #pragma unroll
    for (int j = 0; j < 4; ++j) {
        const int col  = nb + j * 16 + mloc;
        const float bias = b_self[col] + b_neigh[col];
        #pragma unroll
        for (int i = 0; i < 8; ++i) {
            const int m = mb + i + 8 * r;
            outG[(size_t)permS[m] * OUTF + col] = acc[j][i] + bias;
        }
    }
}

// -------------------------------------------------------------------------
extern "C" void kernel_launch(void* const* d_in, const int* in_sizes, int n_in,
                              void* d_out, int out_size, void* d_ws, size_t ws_size,
                              hipStream_t stream) {
    const float* src_feat   = (const float*)d_in[0];
    const float* dst_feat   = (const float*)d_in[1];
    const float* timestamp  = (const float*)d_in[2];
    const float* basis_freq = (const float*)d_in[3];
    const float* phase      = (const float*)d_in[4];
    const float* W_time     = (const float*)d_in[5];
    const float* b_time     = (const float*)d_in[6];
    const float* W_self     = (const float*)d_in[7];
    const float* b_self     = (const float*)d_in[8];
    const float* W_neigh    = (const float*)d_in[9];
    const float* b_neigh    = (const float*)d_in[10];
    const int*   src_perm   = (const int*)d_in[11];
    const int*   dst_perm   = (const int*)d_in[12];
    const int*   src_deg    = (const int*)d_in[13];
    const int*   dst_deg    = (const int*)d_in[14];

    // workspace layout: enc_src | enc_dst (fp32), then hi/lo bf16 weights
    float* enc_src = (float*)d_ws;                      // E*128 f32
    float* enc_dst = enc_src + (size_t)E_EDGES * DIM;   // E*128 f32
    unsigned short* wsb = (unsigned short*)(enc_dst + (size_t)E_EDGES * DIM);
    unsigned short* Wt_hi = wsb;                  // 128*256
    unsigned short* Wt_lo = Wt_hi + DIM * K1;
    unsigned short* Ws_hi = Wt_lo + DIM * K1;     // 128*128
    unsigned short* Ws_lo = Ws_hi + OUTF * DIM;
    unsigned short* Wn_hi = Ws_lo + OUTF * DIM;
    unsigned short* Wn_lo = Wn_hi + OUTF * DIM;

    // split weights (tiny)
    cvt_hilo<<<(DIM * K1 + 255) / 256, 256, 0, stream>>>(W_time,  Wt_hi, Wt_lo, DIM * K1);
    cvt_hilo<<<(OUTF * DIM + 255) / 256, 256, 0, stream>>>(W_self,  Ws_hi, Ws_lo, OUTF * DIM);
    cvt_hilo<<<(OUTF * DIM + 255) / 256, 256, 0, stream>>>(W_neigh, Wn_hi, Wn_lo, OUTF * DIM);

    dim3 g1(E_EDGES / 16, 2);
    tsage_encode<<<g1, 128, 0, stream>>>(
        src_feat, dst_feat, timestamp, basis_freq, phase,
        Wt_hi, Wt_lo, b_time, enc_src, enc_dst);

    dim3 g2(BK, 2);
    tsage_meanconv<<<g2, 128, 0, stream>>>(
        enc_src, enc_dst, src_perm, dst_perm, src_deg, dst_deg,
        Ws_hi, Ws_lo, Wn_hi, Wn_lo, b_self, b_neigh, (float*)d_out);
}
